// SlidingWindowAttention_53283364274704
// MI455X (gfx1250) — matrix operand order repaired and checked by reading, not compiled, not run on hardware
//
#include <hip/hip_runtime.h>

typedef _Float16 v16h __attribute__((ext_vector_type(16)));
typedef _Float16 h2   __attribute__((ext_vector_type(2)));
typedef __fp16   fp16x2 __attribute__((ext_vector_type(2)));
typedef float    v8f  __attribute__((ext_vector_type(8)));

#define SEQ  8192
#define DIM  64
#define WIN  64
#define WAVES_PER_BLOCK 2
#define ROWS_PER_WAVE   16
#define NKEYCHUNKS 5          // 5 * 32 = 160 keys >= 144-key union window of a 16-row tile

union V16 { v16h v; h2 p[8]; };

// v_cvt_pk_rtz_f16_f32 with result re-typed to our _Float16 pair
static __device__ __forceinline__ h2 pk2h(float a, float b)
{
    fp16x2 r = __builtin_amdgcn_cvt_pkrtz(a, b);
    return __builtin_bit_cast(h2, r);
}

// XOR-swap reduction steps via ds_swizzle (group-of-32: imm = xor_mask<<10 | and_mask 0x1F)
#define SWZ_MAXSTEP(x, M) fmaxf((x), __int_as_float(__builtin_amdgcn_ds_swizzle(__float_as_int(x), (0x1F | ((M) << 10)))))
#define SWZ_ADDSTEP(x, M) ((x) + __int_as_float(__builtin_amdgcn_ds_swizzle(__float_as_int(x), (0x1F | ((M) << 10)))))

// Per-lane async DMA: copy 256B (one 64-dim fp32 row) global -> LDS, no VGPR round trip.
// ASYNCcnt += 16. INST_OFFSET is added to BOTH the LDS and global addresses (ISA 10.x async).
__device__ __forceinline__ void stage_v_async(const float* __restrict__ gsrc, float* ldsdst)
{
    unsigned ld = (unsigned)(unsigned long long)ldsdst;          // low 32 bits = LDS byte offset
    unsigned long long gs = (unsigned long long)gsrc;
#define ACP(OFF) asm volatile("global_load_async_to_lds_b128 %0, %1, off offset:" #OFF \
                              :: "v"(ld), "v"(gs) : "memory")
    ACP(0);   ACP(16);  ACP(32);  ACP(48);
    ACP(64);  ACP(80);  ACP(96);  ACP(112);
    ACP(128); ACP(144); ACP(160); ACP(176);
    ACP(192); ACP(208); ACP(224); ACP(240);
#undef ACP
}

__global__ __launch_bounds__(WAVES_PER_BLOCK * 32)
void swa_wmma_kernel(const float* __restrict__ qg,
                     const float* __restrict__ kg,
                     const float* __restrict__ vg,
                     float* __restrict__ outg)
{
    // Per-wave LDS: double-buffered fp32 V chunk (2 x 32 keys x 64 dims) + f16 P tile (16x32)
    __shared__ float    lds_vf[WAVES_PER_BLOCK][2][32 * DIM];   // 32 KB
    __shared__ _Float16 lds_p [WAVES_PER_BLOCK][16 * 32];       //  2 KB

    const int tid  = threadIdx.x;
    const int w    = tid >> 5;
    const int lane = tid & 31;
    const int m    = lane & 15;
    const int hi   = lane >> 4;

    const int blocks_per_batch = SEQ / (WAVES_PER_BLOCK * ROWS_PER_WAVE); // 256
    const int b      = blockIdx.x / blocks_per_batch;
    const int rowblk = (blockIdx.x % blocks_per_batch) * (WAVES_PER_BLOCK * ROWS_PER_WAVE);
    const int q_row0 = rowblk + w * ROWS_PER_WAVE;

    const float scale = 0.125f; // 1/sqrt(64)
    const size_t base = (size_t)b * SEQ * DIM;

    // ---- Q tile -> WMMA A layout, packed f16 pairs via v_cvt_pk_rtz_f16_f32 ----
    // A half j -> K(dim) = 16*hi + j; pairs (2t,2t+1) are consecutive dims.
    const float* qrow = qg + base + (size_t)(q_row0 + m) * DIM;
    V16 a_q0, a_q1;
#pragma unroll
    for (int t = 0; t < 8; ++t) {
        int d = 16 * hi + 2 * t;
        a_q0.p[t] = pk2h(qrow[d],      qrow[d + 1]);
        a_q1.p[t] = pk2h(qrow[32 + d], qrow[32 + d + 1]);
    }

    v8f o0 = {}, o1 = {}, o2 = {}, o3 = {};
    float m_s[8], l_s[8];
#pragma unroll
    for (int g = 0; g < 8; ++g) { m_s[g] = -__builtin_inff(); l_s[g] = 0.0f; }

    _Float16* ldsP = lds_p[w];

    // ---- prologue: async-stage V chunk 0 into buffer 0 ----
    {
        int kr = q_row0 - WIN + lane;
        int kc = kr < 0 ? 0 : (kr >= SEQ ? SEQ - 1 : kr);
        stage_v_async(vg + base + (size_t)kc * DIM, &lds_vf[w][0][lane * DIM]);
    }

    for (int it = 0; it < NKEYCHUNKS; ++it) {
        const int kt0 = q_row0 - WIN + 32 * it;
        const bool more = (it + 1 < NKEYCHUNKS);

        // -- issue next chunk's async V copy into the other buffer (double buffering).
        //    Drain our DS reads first: async LDS writes are unordered vs DS ops.
        if (more) {
            asm volatile("s_wait_dscnt 0" ::: "memory");
            int kr = kt0 + 32 + lane;
            int kc = kr < 0 ? 0 : (kr >= SEQ ? SEQ - 1 : kr);
            stage_v_async(vg + base + (size_t)kc * DIM, &lds_vf[w][(it + 1) & 1][lane * DIM]);
            __builtin_prefetch(kg + base + (size_t)kc * DIM, 0, 1); // next K rows -> cache
        }

        // -- K^T B-tiles straight from global (lane reads 16 consecutive dims of one key
        //    row; B half j -> K(dim) = 16*hi + j, N(key) = m), packed with cvt_pkrtz --
        v8f s0, s1;
        {
            int kk0 = kt0 + m;
            int kk1 = kk0 + 16;
            int kc0 = kk0 < 0 ? 0 : (kk0 >= SEQ ? SEQ - 1 : kk0);
            int kc1 = kk1 < 0 ? 0 : (kk1 >= SEQ ? SEQ - 1 : kk1);
            const float* kr0 = kg + base + (size_t)kc0 * DIM + 16 * hi;
            const float* kr1 = kg + base + (size_t)kc1 * DIM + 16 * hi;
            V16 bk00, bk01, bk10, bk11;
#pragma unroll
            for (int t = 0; t < 8; ++t) {
                bk00.p[t] = pk2h(kr0[2 * t],      kr0[2 * t + 1]);
                bk01.p[t] = pk2h(kr0[32 + 2 * t], kr0[32 + 2 * t + 1]);
                bk10.p[t] = pk2h(kr1[2 * t],      kr1[2 * t + 1]);
                bk11.p[t] = pk2h(kr1[32 + 2 * t], kr1[32 + 2 * t + 1]);
            }
            v8f zc = {};
            s0 = __builtin_amdgcn_wmma_f32_16x16x32_f16(false, a_q0.v, false, bk00.v, (short)0, zc, false, false);
            s0 = __builtin_amdgcn_wmma_f32_16x16x32_f16(false, a_q1.v, false, bk01.v, (short)0, s0, false, false);
            s1 = __builtin_amdgcn_wmma_f32_16x16x32_f16(false, a_q0.v, false, bk10.v, (short)0, zc, false, false);
            s1 = __builtin_amdgcn_wmma_f32_16x16x32_f16(false, a_q1.v, false, bk11.v, (short)0, s1, false, false);
        }

        // -- scale + sliding-window mask (C layout: element (g,lane) = row g+8*hi, col m) --
        float sc0[8], sc1[8];
        const int kcol0 = kt0 + m, kcol1 = kcol0 + 16;
#pragma unroll
        for (int g = 0; g < 8; ++g) {
            int qi = q_row0 + g + 8 * hi;
            int d0 = qi - kcol0; if (d0 < 0) d0 = -d0;
            int d1 = qi - kcol1; if (d1 < 0) d1 = -d1;
            bool ok0 = (kcol0 >= 0) && (kcol0 < SEQ) && (d0 <= WIN);
            bool ok1 = (kcol1 >= 0) && (kcol1 < SEQ) && (d1 <= WIN);
            sc0[g] = ok0 ? s0[g] * scale : -__builtin_inff();
            sc1[g] = ok1 ? s1[g] * scale : -__builtin_inff();
        }

        // -- row max across the 16-lane half via ds_swizzle XOR swaps --
        float rmax[8];
#pragma unroll
        for (int g = 0; g < 8; ++g) {
            float r = fmaxf(sc0[g], sc1[g]);
            r = SWZ_MAXSTEP(r, 1);
            r = SWZ_MAXSTEP(r, 2);
            r = SWZ_MAXSTEP(r, 4);
            r = SWZ_MAXSTEP(r, 8);
            rmax[g] = r;
        }

        // -- online softmax update --
        float p0[8], p1[8], corr[8];
#pragma unroll
        for (int g = 0; g < 8; ++g) {
            float nm = fmaxf(m_s[g], rmax[g]);
            if (nm == -__builtin_inff()) {       // whole chunk invalid for this row
                corr[g] = 1.0f; p0[g] = 0.0f; p1[g] = 0.0f;
            } else {
                corr[g] = __expf(m_s[g] - nm);   // exp(-inf)=0 on first valid chunk
                p0[g]   = __expf(sc0[g] - nm);
                p1[g]   = __expf(sc1[g] - nm);
            }
            m_s[g] = nm;
        }
#pragma unroll
        for (int g = 0; g < 8; ++g) {
            float rs = p0[g] + p1[g];
            rs = SWZ_ADDSTEP(rs, 1);
            rs = SWZ_ADDSTEP(rs, 2);
            rs = SWZ_ADDSTEP(rs, 4);
            rs = SWZ_ADDSTEP(rs, 8);
            l_s[g] = l_s[g] * corr[g] + rs;
            o0[g] *= corr[g]; o1[g] *= corr[g]; o2[g] *= corr[g]; o3[g] *= corr[g];
        }

        // -- P: C-layout -> LDS row-major f16 -> A-layout (transpose through LDS) --
#pragma unroll
        for (int g = 0; g < 8; ++g) {
            int row = g + 8 * hi;
            ldsP[row * 32 + m]      = (_Float16)p0[g];
            ldsP[row * 32 + 16 + m] = (_Float16)p1[g];
        }
        asm volatile("s_wait_dscnt 0" ::: "memory");  // in-wave cross-lane LDS handoff

        V16 pa;
        {
            const h2* pp = (const h2*)ldsP;            // aligned b32 pair loads
#pragma unroll
            for (int t = 0; t < 8; ++t)
                pa.p[t] = pp[m * 16 + 8 * hi + t];
        }

        // -- wait for this chunk's async V copy (16 newer copies may still be in flight;
        //    async loads complete in order) --
        if (more) asm volatile("s_wait_asynccnt 16" ::: "memory");
        else      asm volatile("s_wait_asynccnt 0"  ::: "memory");

        // -- V in B layout from fp32 LDS: half j -> K(key) = j+16*hi, N(dim) = nc*16+m;
        //    convert pairs of keys with cvt_pkrtz --
        const float* vb = lds_vf[w][it & 1];
        V16 bv[4];
#pragma unroll
        for (int t = 0; t < 8; ++t) {
            const float* r0 = vb + (2 * t + 16 * hi) * DIM + m;
            const float* r1 = r0 + DIM;
#pragma unroll
            for (int nc = 0; nc < 4; ++nc)
                bv[nc].p[t] = pk2h(r0[nc * 16], r1[nc * 16]);
        }
        o0 = __builtin_amdgcn_wmma_f32_16x16x32_f16(false, pa.v, false, bv[0].v, (short)0, o0, false, false);
        o1 = __builtin_amdgcn_wmma_f32_16x16x32_f16(false, pa.v, false, bv[1].v, (short)0, o1, false, false);
        o2 = __builtin_amdgcn_wmma_f32_16x16x32_f16(false, pa.v, false, bv[2].v, (short)0, o2, false, false);
        o3 = __builtin_amdgcn_wmma_f32_16x16x32_f16(false, pa.v, false, bv[3].v, (short)0, o3, false, false);
    }

    // ---- normalize and store (lanes 0-15 write consecutive dims: coalesced) ----
#pragma unroll
    for (int g = 0; g < 8; ++g) {
        int row = q_row0 + g + 8 * hi;
        float inv = 1.0f / l_s[g];
        float* orow = outg + base + (size_t)row * DIM + m;
        orow[0]  = o0[g] * inv;
        orow[16] = o1[g] * inv;
        orow[32] = o2[g] * inv;
        orow[48] = o3[g] * inv;
    }
}

extern "C" void kernel_launch(void* const* d_in, const int* in_sizes, int n_in,
                              void* d_out, int out_size, void* d_ws, size_t ws_size,
                              hipStream_t stream)
{
    const float* q = (const float*)d_in[0];
    const float* k = (const float*)d_in[1];
    const float* v = (const float*)d_in[2];
    float* out = (float*)d_out;

    const int B = 2;
    const int nblocks = B * SEQ / (WAVES_PER_BLOCK * ROWS_PER_WAVE); // 512
    dim3 grid(nblocks), block(WAVES_PER_BLOCK * 32);
    hipLaunchKernelGGL(swa_wmma_kernel, grid, block, 0, stream, q, k, v, out);
}
